// SlidingCausalMultiheadAttention_88923002896777
// MI455X (gfx1250) — matrix operand order repaired and checked by reading, not compiled, not run on hardware
//
#include <hip/hip_runtime.h>
#include <hip/hip_bf16.h>
#include <stdint.h>

typedef _Float16 hv16 __attribute__((ext_vector_type(16)));
typedef _Float16 hv8  __attribute__((ext_vector_type(8)));
typedef _Float16 hv4  __attribute__((ext_vector_type(4)));
typedef float    fv8  __attribute__((ext_vector_type(8)));

#define BB 4
#define CC 2
#define TT 4224
#define FF 256
#define LL 256
#define STRIDE_ 128
#define HH 8
#define DH 32
#define NB 32
#define NSEQ 256   /* B*C*NB */

#define AS_LDS __attribute__((address_space(3)))

// ---------------------------------------------------------------------------
// WMMA helpers (gfx1250, wave32). A: 16x32 f16, B: 32x16 f16, C/D: 16x16 f32.
// ---------------------------------------------------------------------------
__device__ __forceinline__ fv8 wmma16(hv16 a, hv16 b, fv8 c) {
  return __builtin_amdgcn_wmma_f32_16x16x32_f16(
      /*neg_a=*/false, a, /*neg_b=*/false, b,
      /*c_mod=*/(short)0, c, /*reuse_a=*/false, /*reuse_b=*/false);
}

// A fragment from row-major [M][K] matrix (ld in halves).
// ISA layout: lanes 0-15 -> M=lane, VGPR0..3 = K 0..7, VGPR4..7 = K 16..23;
//             lanes 16-31 add +8 to K.
__device__ __forceinline__ hv16 load_a_frag(const _Float16* m, int ld,
                                            int row0, int k0, int lane) {
  int g = lane >> 4, r = lane & 15;
  const _Float16* p = m + (size_t)(row0 + r) * ld + k0 + g * 8;
  hv8 lo = *(const hv8*)(p);        // K = g*8 .. g*8+7
  hv8 hi = *(const hv8*)(p + 16);   // K = 16+g*8 .. 16+g*8+7
  hv16 out;
#pragma unroll
  for (int t = 0; t < 8; ++t) { out[t] = lo[t]; out[t + 8] = hi[t]; }
  return out;
}

// B fragment from N-major storage Bt[N][K] (i.e. B^T row-major; ld in halves).
// ISA layout: lanes 0-15 -> N=lane, VGPR0..3 = K 0..7, VGPR4..7 = K 16..23;
//             lanes 16-31 add +8 to K.
__device__ __forceinline__ hv16 load_b_frag(const _Float16* bt, int ld,
                                            int n0, int k0, int lane) {
  int g = lane >> 4, n = lane & 15;
  const _Float16* p = bt + (size_t)(n0 + n) * ld + k0 + g * 8;
  hv8 lo = *(const hv8*)(p);
  hv8 hi = *(const hv8*)(p + 16);
  hv16 out;
#pragma unroll
  for (int t = 0; t < 8; ++t) { out[t] = lo[t]; out[t + 8] = hi[t]; }
  return out;
}

// ---------------------------------------------------------------------------
// CDNA5 async global -> LDS copy (16 bytes per lane), tracked by ASYNCcnt.
// Inline asm (portable across toolchains; builtin signatures differ).
// vDst = 32-bit LDS byte offset, address = 64-bit global VA in a VGPR pair.
// ---------------------------------------------------------------------------
__device__ __forceinline__ void async_cp16(const _Float16* gsrc, _Float16* ldst) {
  unsigned lds_off = (unsigned)(uintptr_t)(AS_LDS void*)ldst;
  unsigned long long ga = (unsigned long long)(uintptr_t)gsrc;
  asm volatile("global_load_async_to_lds_b128 %0, %1, off"
               :: "v"(lds_off), "v"(ga)
               : "memory");
}
__device__ __forceinline__ void async_wait0() {
  asm volatile("s_wait_asynccnt 0" ::: "memory");
}

// ---------------------------------------------------------------------------
// Kernel 0: convert weights fp32 -> f16
// ---------------------------------------------------------------------------
__global__ __launch_bounds__(256) void conv_weights_k(
    const float* __restrict__ w_in, const float* __restrict__ w_out,
    _Float16* __restrict__ wf_in, _Float16* __restrict__ wf_out) {
  int i = blockIdx.x * 256 + threadIdx.x;
  if (i < 3 * FF * FF) wf_in[i] = (_Float16)w_in[i];
  if (i < FF * FF)     wf_out[i] = (_Float16)w_out[i];
}

// ---------------------------------------------------------------------------
// Kernel 1: QKV projection. 128 token-rows per WG; 8 waves = 8 row tiles.
// q/k layout: [n][h][i][d]; v layout (transposed): [n][h][d][i].
// Per column tile: clause-load all 8 B frags, then 8 WMMAs on 2 acc chains.
// ---------------------------------------------------------------------------
#define XS_LD 264
__global__ __launch_bounds__(256) void qkv_k(
    const float* __restrict__ x, const _Float16* __restrict__ wf_in,
    const float* __restrict__ b_in,
    _Float16* __restrict__ q_ws, _Float16* __restrict__ k_ws,
    _Float16* __restrict__ v_ws) {
  extern __shared__ char smem[];
  _Float16* Xs = (_Float16*)smem;            // [128][XS_LD]
  int tid = threadIdx.x;
  int base = blockIdx.x * 128;

  // gather x rows -> f16 LDS tile
  for (int e = tid; e < 128 * 64; e += 256) {
    int r = e >> 6; int c4 = (e & 63) << 2;
    int tok = base + r; int n = tok >> 8; int i = tok & 255;
    int blk = n & 31; int bc = n >> 5;
    const float4 v = *(const float4*)(x + ((size_t)(bc * TT + blk * STRIDE_ + i)) * FF + c4);
    hv4 h = { (_Float16)v.x, (_Float16)v.y, (_Float16)v.z, (_Float16)v.w };
    *(hv4*)(Xs + r * XS_LD + c4) = h;
  }
  __syncthreads();

  int wave = tid >> 5, lane = tid & 31, g = lane >> 4, nn = lane & 15;
  hv16 afr[8];
#pragma unroll
  for (int kt = 0; kt < 8; ++kt)
    afr[kt] = load_a_frag(Xs, XS_LD, wave * 16, kt * 32, lane);

  // ---- q tiles: ct 0..15
  for (int ct = 0; ct < 16; ++ct) {
    hv16 bf[8];
#pragma unroll
    for (int kt = 0; kt < 8; ++kt)
      bf[kt] = load_b_frag(wf_in, FF, ct * 16, kt * 32, lane);
    fv8 a0 = {}, a1 = {};
#pragma unroll
    for (int kt = 0; kt < 8; kt += 2) {
      a0 = wmma16(afr[kt], bf[kt], a0);
      a1 = wmma16(afr[kt + 1], bf[kt + 1], a1);
    }
    int f = ct * 16 + nn;
    float bias = b_in[f];
    int hh = f >> 5, d = f & 31;
#pragma unroll
    for (int e = 0; e < 8; ++e) {
      int tok = base + wave * 16 + e + 8 * g;
      int n = tok >> 8, i = tok & 255;
      q_ws[(((size_t)n * HH + hh) * LL + i) * DH + d] = (_Float16)(a0[e] + a1[e] + bias);
    }
  }
  // ---- k tiles: ct 16..31
  for (int ct = 16; ct < 32; ++ct) {
    hv16 bf[8];
#pragma unroll
    for (int kt = 0; kt < 8; ++kt)
      bf[kt] = load_b_frag(wf_in, FF, ct * 16, kt * 32, lane);
    fv8 a0 = {}, a1 = {};
#pragma unroll
    for (int kt = 0; kt < 8; kt += 2) {
      a0 = wmma16(afr[kt], bf[kt], a0);
      a1 = wmma16(afr[kt + 1], bf[kt + 1], a1);
    }
    int col = ct * 16 + nn;
    float bias = b_in[col];
    int f = col - FF, hh = f >> 5, d = f & 31;
#pragma unroll
    for (int e = 0; e < 8; ++e) {
      int tok = base + wave * 16 + e + 8 * g;
      int n = tok >> 8, i = tok & 255;
      k_ws[(((size_t)n * HH + hh) * LL + i) * DH + d] = (_Float16)(a0[e] + a1[e] + bias);
    }
  }
  // ---- v tiles: ct 32..47 (stored transposed [n][h][d][i])
  for (int ct = 32; ct < 48; ++ct) {
    hv16 bf[8];
#pragma unroll
    for (int kt = 0; kt < 8; ++kt)
      bf[kt] = load_b_frag(wf_in, FF, ct * 16, kt * 32, lane);
    fv8 a0 = {}, a1 = {};
#pragma unroll
    for (int kt = 0; kt < 8; kt += 2) {
      a0 = wmma16(afr[kt], bf[kt], a0);
      a1 = wmma16(afr[kt + 1], bf[kt + 1], a1);
    }
    int col = ct * 16 + nn;
    float bias = b_in[col];
    int f = col - 2 * FF, hh = f >> 5, d = f & 31;
#pragma unroll
    for (int e = 0; e < 8; ++e) {
      int tok = base + wave * 16 + e + 8 * g;
      int n = tok >> 8, i = tok & 255;
      v_ws[(((size_t)n * HH + hh) * DH + d) * LL + i] = (_Float16)(a0[e] + a1[e] + bias);
    }
  }
}

// ---------------------------------------------------------------------------
// Kernel 2: attention per (n,h). 8 waves; wave w owns query rows w*32..+31.
// scores = (Q K^T)*scale + band(1.0), full softmax, O = P V.
// Q/K/V staged via CDNA5 async global->LDS copies (ASYNCcnt).
// ---------------------------------------------------------------------------
#define QK_LD 40
#define SS_LD 264
__global__ __launch_bounds__(256) void attn_k(
    const _Float16* __restrict__ q_ws, const _Float16* __restrict__ k_ws,
    const _Float16* __restrict__ v_ws, _Float16* __restrict__ o_ws) {
  extern __shared__ char smem[];
  _Float16* Qs = (_Float16*)smem;            // [256][QK_LD]
  _Float16* Ks = Qs + LL * QK_LD;            // [256][QK_LD]
  _Float16* Vt = Ks + LL * QK_LD;            // [32][SS_LD] (d-major)
  _Float16* Ss = Vt + DH * SS_LD;            // [256][SS_LD]

  int tid = threadIdx.x;
  int n = blockIdx.x >> 3, h = blockIdx.x & 7;
  size_t nh = (size_t)n * HH + h;
  const _Float16* qg = q_ws + nh * LL * DH;
  const _Float16* kg = k_ws + nh * LL * DH;
  const _Float16* vg = v_ws + nh * DH * LL;

  for (int e = tid; e < LL * 4; e += 256) {    // Q,K: 256 rows x 4 chunks of 8
    int r = e >> 2; int c8 = (e & 3) * 8;
    async_cp16(qg + r * DH + c8, Qs + r * QK_LD + c8);
    async_cp16(kg + r * DH + c8, Ks + r * QK_LD + c8);
  }
  for (int e = tid; e < DH * 32; e += 256) {   // Vt: 32 rows x 32 chunks of 8
    int d = e >> 5; int c8 = (e & 31) * 8;
    async_cp16(vg + d * LL + c8, Vt + d * SS_LD + c8);
  }
  async_wait0();
  __syncthreads();

  int wave = tid >> 5, lane = tid & 31, g = lane >> 4, nn = lane & 15;
  int r0 = wave * 32;
  const float scale = 0.17677669529663687f;   // 1/sqrt(32)

  // ---- scores: dh=32 => single WMMA per 16x16 tile; 2 j-tiles per step
  hv16 aq0 = load_a_frag(Qs, QK_LD, r0, 0, lane);
  hv16 aq1 = load_a_frag(Qs, QK_LD, r0 + 16, 0, lane);
  for (int jt = 0; jt < 16; jt += 2) {
    hv16 b0 = load_b_frag(Ks, QK_LD, jt * 16, 0, lane);
    hv16 b1 = load_b_frag(Ks, QK_LD, jt * 16 + 16, 0, lane);
    fv8 c00 = {}, c10 = {}, c01 = {}, c11 = {};
    c00 = wmma16(aq0, b0, c00);
    c10 = wmma16(aq1, b0, c10);
    c01 = wmma16(aq0, b1, c01);
    c11 = wmma16(aq1, b1, c11);
    int j0 = jt * 16 + nn, j1 = j0 + 16;
#pragma unroll
    for (int e = 0; e < 8; ++e) {
      int M = e + 8 * g;
      int i0 = r0 + M, i1 = r0 + 16 + M;
      Ss[i0 * SS_LD + j0] = (_Float16)(c00[e] * scale + ((i0 >= j0 && (i0 - j0) < 128) ? 1.0f : 0.0f));
      Ss[i1 * SS_LD + j0] = (_Float16)(c10[e] * scale + ((i1 >= j0 && (i1 - j0) < 128) ? 1.0f : 0.0f));
      Ss[i0 * SS_LD + j1] = (_Float16)(c01[e] * scale + ((i0 >= j1 && (i0 - j1) < 128) ? 1.0f : 0.0f));
      Ss[i1 * SS_LD + j1] = (_Float16)(c11[e] * scale + ((i1 >= j1 && (i1 - j1) < 128) ? 1.0f : 0.0f));
    }
  }
  __syncthreads();

  // ---- softmax: one row per thread
  {
    _Float16* srow = Ss + tid * SS_LD;
    float mx = -1e30f;
    for (int j = 0; j < LL; ++j) { float v = (float)srow[j]; mx = fmaxf(mx, v); }
    float sum = 0.0f;
    for (int j = 0; j < LL; ++j) {
      float ev = __expf((float)srow[j] - mx);
      sum += ev;
      srow[j] = (_Float16)ev;
    }
    float inv = 1.0f / sum;
    for (int j = 0; j < LL; ++j)
      srow[j] = (_Float16)((float)srow[j] * inv);
  }
  __syncthreads();

  // ---- O = P V  (2 row tiles x 2 dh tiles, K loop over 256 tokens)
  fv8 o00 = {}, o01 = {}, o10 = {}, o11 = {};
#pragma unroll 2
  for (int jt = 0; jt < 8; ++jt) {
    hv16 a0 = load_a_frag(Ss, SS_LD, r0, jt * 32, lane);
    hv16 a1 = load_a_frag(Ss, SS_LD, r0 + 16, jt * 32, lane);
    hv16 b0 = load_b_frag(Vt, SS_LD, 0, jt * 32, lane);
    hv16 b1 = load_b_frag(Vt, SS_LD, 16, jt * 32, lane);
    o00 = wmma16(a0, b0, o00); o01 = wmma16(a0, b1, o01);
    o10 = wmma16(a1, b0, o10); o11 = wmma16(a1, b1, o11);
  }
  _Float16* og = o_ws + (size_t)n * LL * FF;   // o_ws[(n*L + i)*F + h*32 + d]
#pragma unroll
  for (int e = 0; e < 8; ++e) {
    int M = e + 8 * g;
    int i0 = r0 + M, i1 = r0 + 16 + M;
    og[(size_t)i0 * FF + h * DH + nn]      = (_Float16)o00[e];
    og[(size_t)i0 * FF + h * DH + 16 + nn] = (_Float16)o01[e];
    og[(size_t)i1 * FF + h * DH + nn]      = (_Float16)o10[e];
    og[(size_t)i1 * FF + h * DH + 16 + nn] = (_Float16)o11[e];
  }
}

// ---------------------------------------------------------------------------
// Kernel 3: overlap-average (<=2 covering blocks) + output projection + bias.
// 128 output-token rows per WG over B*C*T = 33792 tokens.
// ---------------------------------------------------------------------------
__global__ __launch_bounds__(256) void outproj_k(
    const _Float16* __restrict__ o_ws, const _Float16* __restrict__ wf_out,
    const float* __restrict__ b_out, float* __restrict__ y) {
  extern __shared__ char smem[];
  _Float16* Xs = (_Float16*)smem;            // [128][XS_LD]
  int tid = threadIdx.x;
  int base = blockIdx.x * 128;

  for (int e = tid; e < 128 * 32; e += 256) {
    int r = e >> 5; int c8 = (e & 31) << 3;
    int gtok = base + r;
    int bc = gtok / TT; int t = gtok % TT;
    int lo = (t >= LL) ? (((t - LL) >> 7) + 1) : 0;
    int hi = t >> 7; if (hi > NB - 1) hi = NB - 1;
    int n0 = bc * NB + lo; int i0 = t - lo * STRIDE_;
    hv8 h0 = *(const hv8*)(o_ws + ((size_t)n0 * LL + i0) * FF + c8);
    hv8 outv;
    if (hi > lo) {
      hv8 h1 = *(const hv8*)(o_ws + ((size_t)(n0 + 1) * LL + (i0 - STRIDE_)) * FF + c8);
#pragma unroll
      for (int t2 = 0; t2 < 8; ++t2)
        outv[t2] = (_Float16)(((float)h0[t2] + (float)h1[t2]) * 0.5f);
    } else {
      outv = h0;
    }
    *(hv8*)(Xs + r * XS_LD + c8) = outv;
  }
  __syncthreads();

  int wave = tid >> 5, lane = tid & 31, g = lane >> 4, nn = lane & 15;
  hv16 afr[8];
#pragma unroll
  for (int kt = 0; kt < 8; ++kt)
    afr[kt] = load_a_frag(Xs, XS_LD, wave * 16, kt * 32, lane);

  for (int ct = 0; ct < 16; ++ct) {
    hv16 bf[8];
#pragma unroll
    for (int kt = 0; kt < 8; ++kt)
      bf[kt] = load_b_frag(wf_out, FF, ct * 16, kt * 32, lane);
    fv8 a0 = {}, a1 = {};
#pragma unroll
    for (int kt = 0; kt < 8; kt += 2) {
      a0 = wmma16(afr[kt], bf[kt], a0);
      a1 = wmma16(afr[kt + 1], bf[kt + 1], a1);
    }
    int col = ct * 16 + nn;
    float bias = b_out[col];
#pragma unroll
    for (int e = 0; e < 8; ++e) {
      int gtok = base + wave * 16 + e + 8 * g;
      y[(size_t)gtok * FF + col] = a0[e] + a1[e] + bias;
    }
  }
}

// ---------------------------------------------------------------------------
extern "C" void kernel_launch(void* const* d_in, const int* in_sizes, int n_in,
                              void* d_out, int out_size, void* d_ws, size_t ws_size,
                              hipStream_t stream) {
  const float* x     = (const float*)d_in[0];
  const float* w_in  = (const float*)d_in[1];
  const float* b_in  = (const float*)d_in[2];
  const float* w_out = (const float*)d_in[3];
  const float* b_out = (const float*)d_in[4];
  float* y = (float*)d_out;

  char* ws = (char*)d_ws;
  const size_t QKV_ELEMS = (size_t)NSEQ * HH * LL * DH;   // 16,777,216
  _Float16* wf_in  = (_Float16*)ws;                        // 3*F*F halves
  _Float16* wf_out = (_Float16*)(ws + (size_t)3 * FF * FF * 2);
  _Float16* q_ws   = (_Float16*)(ws + 524288);
  _Float16* k_ws   = q_ws + QKV_ELEMS;
  _Float16* v_ws   = k_ws + QKV_ELEMS;
  _Float16* o_ws   = v_ws + QKV_ELEMS;

  const int SM_QKV  = 128 * XS_LD * 2;                                 // 67584
  const int SM_ATTN = (LL * QK_LD * 2 + DH * SS_LD + LL * SS_LD) * 2;  // 193024

  (void)hipFuncSetAttribute((const void*)qkv_k,
      hipFuncAttributeMaxDynamicSharedMemorySize, SM_QKV);
  (void)hipFuncSetAttribute((const void*)attn_k,
      hipFuncAttributeMaxDynamicSharedMemorySize, SM_ATTN);
  (void)hipFuncSetAttribute((const void*)outproj_k,
      hipFuncAttributeMaxDynamicSharedMemorySize, SM_QKV);

  conv_weights_k<<<768, 256, 0, stream>>>(w_in, w_out, wf_in, wf_out);
  qkv_k<<<512, 256, SM_QKV, stream>>>(x, wf_in, b_in, q_ws, k_ws, v_ws);
  attn_k<<<NSEQ * HH, 256, SM_ATTN, stream>>>(q_ws, k_ws, v_ws, o_ws);
  outproj_k<<<(BB * CC * TT) / 128, 256, SM_QKV, stream>>>(o_ws, wf_out, b_out, y);
}
